// Transformer2Tests_37391985279390
// MI455X (gfx1250) — matrix operand order repaired and checked by reading, not compiled, not run on hardware
//
#include <hip/hip_runtime.h>

#define Hd   1024
#define Fd   4096
#define Bd   4
#define NHd  16
#define DHd  64

typedef __attribute__((ext_vector_type(16))) __bf16 v16bf;
typedef __attribute__((ext_vector_type(8)))  float  v8f;

__device__ inline __bf16 f2bf(float f) {
  unsigned u = __float_as_uint(f);
  u += 0x7FFFu + ((u >> 16) & 1u);               // round-to-nearest-even
  unsigned short s = (unsigned short)(u >> 16);
  __bf16 b;
  __builtin_memcpy(&b, &s, 2);
  return b;
}

__device__ inline v8f wmma_bf16(v16bf a, v16bf b, v8f c) {
  return __builtin_amdgcn_wmma_f32_16x16x32_bf16(false, a, false, b, (short)0, c,
                                                 false, false);
}

__device__ inline v8f vzero8() {
  v8f z;
#pragma unroll
  for (int i = 0; i < 8; ++i) z[i] = 0.f;
  return z;
}

// A-operand (16x32 bf16), row-major LDS tile, ld must be a multiple of 8 so the
// two 8-element runs are 16B-aligned contiguous -> 2x ds_load_b128 per fragment.
__device__ inline v16bf lds_a_frag(const __bf16* base, int ld, int row0, int k0, int lane) {
  v16bf a;
  const int row = row0 + (lane & 15);
  const int kb0 = k0 + ((lane & 16) ? 16 : 0);
  const int kb1 = kb0 + 8;
  const __bf16* p0 = base + row * ld + kb0;
  const __bf16* p1 = base + row * ld + kb1;
#pragma unroll
  for (int e = 0; e < 8; ++e) { a[e] = p0[e]; a[e + 8] = p1[e]; }
  return a;
}

// B-operand (32x16 bf16) from an LDS tile stored FRAGMENT-MAJOR: [col][k]
// (K contiguous per output column). 16 contiguous bf16 -> 2x ds_load_b128.
__device__ inline v16bf lds_b_frag(const __bf16* base, int ld, int col0, int k0, int lane) {
  v16bf b;
  const int col = col0 + (lane & 15);
  const int ks  = k0 + ((lane & 16) ? 16 : 0);
  const __bf16* p = base + col * ld + ks;
#pragma unroll
  for (int e = 0; e < 16; ++e) b[e] = p[e];
  return b;
}

// ---------------------------------------------------------------------------
// Tile movers for the pipelined GEMM (128x128x32 block, 8 waves as 2x4).
// ---------------------------------------------------------------------------
__device__ inline void ld_tA(const float* A, int row0, int K, int k0, int tid, float4* fa) {
#pragma unroll
  for (int j = 0; j < 4; ++j) {
    const int i = tid + j * 256;
    const int r = i >> 3, c = (i & 7) << 2;
    fa[j] = *(const float4*)(A + (size_t)(row0 + r) * K + k0 + c);
  }
}
__device__ inline void ld_tB(const float* W, int col0, int N, int k0, int tid, float4* fb) {
#pragma unroll
  for (int j = 0; j < 4; ++j) {
    const int i = tid + j * 256;
    const int r = i >> 5, c = (i & 31) << 2;
    fb[j] = *(const float4*)(W + (size_t)(k0 + r) * N + col0 + c);
  }
}
__device__ inline void st_tA(__bf16 (*As)[40], int tid, const float4* fa) {
#pragma unroll
  for (int j = 0; j < 4; ++j) {
    const int i = tid + j * 256;
    const int r = i >> 3, c = (i & 7) << 2;
    As[r][c]     = f2bf(fa[j].x); As[r][c + 1] = f2bf(fa[j].y);
    As[r][c + 2] = f2bf(fa[j].z); As[r][c + 3] = f2bf(fa[j].w);
  }
}
__device__ inline void st_tB(__bf16 (*BsT)[40], int tid, const float4* fb) {
#pragma unroll
  for (int j = 0; j < 4; ++j) {               // transpose: BsT[n][k]
    const int i = tid + j * 256;
    const int r = i >> 5, c = (i & 31) << 2;
    BsT[c + 0][r] = f2bf(fb[j].x);
    BsT[c + 1][r] = f2bf(fb[j].y);
    BsT[c + 2][r] = f2bf(fb[j].z);
    BsT[c + 3][r] = f2bf(fb[j].w);
  }
}

// ---------------------------------------------------------------------------
// GEMM: C = act(A[M,K] @ W[K,N] (+bias) (+R)). M%128==0, N%128==0, K%32==0.
// Software-pipelined: next K-slab fetched to registers during WMMA.
// ---------------------------------------------------------------------------
template <bool HASB, bool HASR, int RELU>
__global__ __launch_bounds__(256) void k_gemm_t(
    const float* __restrict__ A, const float* __restrict__ W,
    const float* __restrict__ bias, const float* __restrict__ R,
    float* __restrict__ C, int M, int N, int K)
{
  __shared__ __bf16 As[128][40];    // [m][k]
  __shared__ __bf16 BsT[128][40];   // [n][k]  fragment-major
  const int tid = threadIdx.x, lane = tid & 31, wave = tid >> 5;
  const int wr = wave >> 2, wc = wave & 3;
  const int row0 = blockIdx.y * 128, col0 = blockIdx.x * 128;

  v8f acc[4][2];
#pragma unroll
  for (int m = 0; m < 4; ++m)
#pragma unroll
    for (int n = 0; n < 2; ++n) acc[m][n] = vzero8();

  float4 fa[4], fb[4];
  ld_tA(A, row0, K, 0, tid, fa);
  ld_tB(W, col0, N, 0, tid, fb);
  st_tA(As, tid, fa);
  st_tB(BsT, tid, fb);
  __syncthreads();

  for (int k0 = 0; k0 < K; k0 += 32) {
    const bool more = (k0 + 32) < K;
    if (more) {
      __builtin_prefetch(A + (size_t)(row0 + (tid >> 1)) * K + k0 + 64, 0, 1);
      ld_tA(A, row0, K, k0 + 32, tid, fa);
      ld_tB(W, col0, N, k0 + 32, tid, fb);
    }
    v16bf af[4], bfr[2];
#pragma unroll
    for (int m = 0; m < 4; ++m) af[m] = lds_a_frag(&As[0][0], 40, wr * 64 + m * 16, 0, lane);
#pragma unroll
    for (int n = 0; n < 2; ++n) bfr[n] = lds_b_frag(&BsT[0][0], 40, wc * 32 + n * 16, 0, lane);
#pragma unroll
    for (int m = 0; m < 4; ++m)
#pragma unroll
      for (int n = 0; n < 2; ++n) acc[m][n] = wmma_bf16(af[m], bfr[n], acc[m][n]);
    __syncthreads();
    if (more) { st_tA(As, tid, fa); st_tB(BsT, tid, fb); }
    __syncthreads();
  }

  const int cl = lane & 15;
  const int rh = (lane & 16) ? 8 : 0;
#pragma unroll
  for (int m = 0; m < 4; ++m)
#pragma unroll
    for (int n = 0; n < 2; ++n) {
      const int col = col0 + wc * 32 + n * 16 + cl;
      float bv = 0.f;
      if (HASB) bv = bias[col];
#pragma unroll
      for (int v = 0; v < 8; ++v) {
        const int row = row0 + wr * 64 + m * 16 + rh + v;
        float val = acc[m][n][v] + bv;
        if (HASR) val += R[(size_t)row * N + col];
        if (RELU) val = fmaxf(val, 0.f);
        C[(size_t)row * N + col] = val;
      }
    }
}

// ---------------------------------------------------------------------------
// Tapped GEMM: conv1d(k=3,p=1)/ConvTranspose1d(k=3,p=1,op=s-1) + ReLU.
// MODE 0: in (B,Cin,Tin), w (Cout,Cin,3)          [embedding]
// MODE 1: in (B,Tin,Cin), w (Cout,Cin,3)          [conv0]
// MODE 2: deconv, in (B,Tin,Cin), w (Cin,Cout,3)  [decoder]
// out (B,Tout,Cout), Tout = 1<<tsh. Grid (Cout/128, B*Tout/128).
// ---------------------------------------------------------------------------
template <int MODE>
__global__ __launch_bounds__(256) void k_conv3_t(
    const float* __restrict__ in, const float* __restrict__ w,
    const float* __restrict__ bias, float* __restrict__ out,
    int Tin, int tsh, int Cin, int Cout, int stride)
{
  __shared__ __bf16 As[128][40];
  __shared__ __bf16 BsT[128][40];
  const int tid = threadIdx.x, lane = tid & 31, wave = tid >> 5;
  const int wr = wave >> 2, wc = wave & 3;
  const int row0 = blockIdx.y * 128, col0 = blockIdx.x * 128;
  const int tmask = (1 << tsh) - 1;

  v8f acc[4][2];
#pragma unroll
  for (int m = 0; m < 4; ++m)
#pragma unroll
    for (int n = 0; n < 2; ++n) acc[m][n] = vzero8();

  for (int tap = 0; tap < 3; ++tap) {
    for (int c0 = 0; c0 < Cin; c0 += 32) {
      for (int i = tid; i < 4096; i += 256) {        // A tile 128x32 (gathered)
        const int r = i >> 5, cc = i & 31;
        const int grow = row0 + r;
        const int b = grow >> tsh, tp = grow & tmask;
        const int c = c0 + cc;
        float val = 0.f;
        if (MODE == 2) {
          const int num = tp + 1 - tap;
          if (num >= 0 && (num & (stride - 1)) == 0) {
            const int t = num >> (stride - 1);
            if (t < Tin) val = in[((size_t)b * Tin + t) * Cin + c];
          }
        } else {
          const int t = tp + tap - 1;
          if (t >= 0 && t < Tin)
            val = (MODE == 0) ? in[((size_t)b * Cin + c) * Tin + t]
                              : in[((size_t)b * Tin + t) * Cin + c];
        }
        As[r][cc] = f2bf(val);
      }
      for (int i = tid; i < 4096; i += 256) {        // B tile -> [n][k]
        const int r = i >> 7, cc = i & 127;
        const int hcol = col0 + cc;
        const int c = c0 + r;
        const float wv = (MODE == 2)
            ? w[((size_t)c * Cout + hcol) * 3 + tap]
            : w[((size_t)hcol * Cin + c) * 3 + tap];
        BsT[cc][r] = f2bf(wv);
      }
      __syncthreads();
      v16bf af[4], bfr[2];
#pragma unroll
      for (int m = 0; m < 4; ++m) af[m] = lds_a_frag(&As[0][0], 40, wr * 64 + m * 16, 0, lane);
#pragma unroll
      for (int n = 0; n < 2; ++n) bfr[n] = lds_b_frag(&BsT[0][0], 40, wc * 32 + n * 16, 0, lane);
#pragma unroll
      for (int m = 0; m < 4; ++m)
#pragma unroll
        for (int n = 0; n < 2; ++n) acc[m][n] = wmma_bf16(af[m], bfr[n], acc[m][n]);
      __syncthreads();
    }
  }
  const int cl = lane & 15;
  const int rh = (lane & 16) ? 8 : 0;
#pragma unroll
  for (int m = 0; m < 4; ++m)
#pragma unroll
    for (int n = 0; n < 2; ++n) {
      const int col = col0 + wc * 32 + n * 16 + cl;
      const float bv = bias[col];
#pragma unroll
      for (int v = 0; v < 8; ++v) {
        const int row = row0 + wr * 64 + m * 16 + rh + v;
        out[(size_t)row * Cout + col] = fmaxf(acc[m][n][v] + bv, 0.f);
      }
    }
}

// ---------------------------------------------------------------------------
// Fused flash attention. Grid (Tq/128, NH, B). DH=64.
// local=1: block-diagonal MS=128 mask -> only the diagonal key block.
// LDS: R1 = Q tile (only until fragments are in regs) then P matrix [128][136];
//      R2 = K tile [key][d] (ld 72) then V transposed [d][key] (ld 136).
// All strides 16B-multiples -> every fragment is ds_load_b128s.
// ---------------------------------------------------------------------------
__global__ __launch_bounds__(256) void k_attn(
    const float* __restrict__ Qg, const float* __restrict__ Kg,
    const float* __restrict__ Vg, float* __restrict__ Og,
    int Tq, int Tk, float scale, int local)
{
  __shared__ __bf16 R1[128 * 136];   // Q tile / P matrix
  __shared__ __bf16 R2[128 * 72];    // K tile / V transposed (64*136 <= 128*72)
  const int tid  = threadIdx.x;
  const int lane = tid & 31;
  const int wave = tid >> 5;
  const int qb   = blockIdx.x;
  const int hoff = blockIdx.y * DHd;
  const int b    = blockIdx.z;
  const float* Qp = Qg + (size_t)b * Tq * Hd + hoff;
  const float* Kp = Kg + (size_t)b * Tk * Hd + hoff;
  const float* Vp = Vg + (size_t)b * Tk * Hd + hoff;
  float*       Op = Og + (size_t)b * Tq * Hd + hoff;

  for (int i = tid; i < 2048; i += 256) {            // Q tile 128x64 -> R1 (ld 136)
    const int r = i >> 4, c = (i & 15) << 2;
    const float4 f = *(const float4*)(Qp + (size_t)(qb * 128 + r) * Hd + c);
    R1[r * 136 + c]     = f2bf(f.x); R1[r * 136 + c + 1] = f2bf(f.y);
    R1[r * 136 + c + 2] = f2bf(f.z); R1[r * 136 + c + 3] = f2bf(f.w);
  }
  __syncthreads();

  const int wq0 = wave * 16;                         // each wave owns 16 q rows
  const v16bf qf0 = lds_a_frag(R1, 136, wq0, 0, lane);
  const v16bf qf1 = lds_a_frag(R1, 136, wq0, 32, lane);

  v8f oacc[4];
#pragma unroll
  for (int n = 0; n < 4; ++n) oacc[n] = vzero8();
  float mrow[8], lrow[8];
#pragma unroll
  for (int v = 0; v < 8; ++v) { mrow[v] = -3.0e38f; lrow[v] = 0.f; }

  const int kbeg = local ? qb : 0;
  const int kend = local ? (qb + 1) : (Tk >> 7);
  for (int kb = kbeg; kb < kend; ++kb) {
    __syncthreads();                                 // prior uses of R1/R2 done
    for (int i = tid; i < 2048; i += 256) {          // K tile 128x64 -> R2 [key][d]
      const int r = i >> 4, c = (i & 15) << 2;
      const float4 f = *(const float4*)(Kp + (size_t)(kb * 128 + r) * Hd + c);
      R2[r * 72 + c]     = f2bf(f.x); R2[r * 72 + c + 1] = f2bf(f.y);
      R2[r * 72 + c + 2] = f2bf(f.z); R2[r * 72 + c + 3] = f2bf(f.w);
    }
    __syncthreads();
    v8f sacc[8];
#pragma unroll
    for (int n = 0; n < 8; ++n) sacc[n] = vzero8();
#pragma unroll
    for (int nt = 0; nt < 8; ++nt) {                 // S = Q @ K^T
      const v16bf b0 = lds_b_frag(R2, 72, nt * 16, 0, lane);
      const v16bf b1 = lds_b_frag(R2, 72, nt * 16, 32, lane);
      sacc[nt] = wmma_bf16(qf0, b0, sacc[nt]);
      sacc[nt] = wmma_bf16(qf1, b1, sacc[nt]);
    }
    __syncthreads();                                 // K reads done; reuse R1->P, R2->Vt
#pragma unroll
    for (int v = 0; v < 8; ++v) {                    // online softmax per q-row
      float mx = -3.0e38f;
#pragma unroll
      for (int nt = 0; nt < 8; ++nt) mx = fmaxf(mx, sacc[nt][v] * scale);
      mx = fmaxf(mx, __shfl_xor(mx, 1, 32));
      mx = fmaxf(mx, __shfl_xor(mx, 2, 32));
      mx = fmaxf(mx, __shfl_xor(mx, 4, 32));
      mx = fmaxf(mx, __shfl_xor(mx, 8, 32));
      const float mnew  = fmaxf(mrow[v], mx);
      const float alpha = __expf(mrow[v] - mnew);
      const int prow = wq0 + v + ((lane & 16) ? 8 : 0);
      float rs = 0.f;
#pragma unroll
      for (int nt = 0; nt < 8; ++nt) {
        const float p = __expf(sacc[nt][v] * scale - mnew);
        rs += p;
        R1[prow * 136 + nt * 16 + (lane & 15)] = f2bf(p);
      }
      rs += __shfl_xor(rs, 1, 32);
      rs += __shfl_xor(rs, 2, 32);
      rs += __shfl_xor(rs, 4, 32);
      rs += __shfl_xor(rs, 8, 32);
      lrow[v] = lrow[v] * alpha + rs;
      mrow[v] = mnew;
#pragma unroll
      for (int nt = 0; nt < 4; ++nt) oacc[nt][v] *= alpha;
    }
    for (int i = tid; i < 2048; i += 256) {          // V tile -> R2 transposed [d][key]
      const int key = i >> 4, d = (i & 15) << 2;
      const float4 f = *(const float4*)(Vp + (size_t)(kb * 128 + key) * Hd + d);
      R2[(d + 0) * 136 + key] = f2bf(f.x);
      R2[(d + 1) * 136 + key] = f2bf(f.y);
      R2[(d + 2) * 136 + key] = f2bf(f.z);
      R2[(d + 3) * 136 + key] = f2bf(f.w);
    }
    __syncthreads();                                 // Vt visible to all waves
#pragma unroll
    for (int kc = 0; kc < 4; ++kc) {                 // O += P @ V
      const v16bf pf = lds_a_frag(R1, 136, wq0, kc * 32, lane);
#pragma unroll
      for (int nt = 0; nt < 4; ++nt) {
        const v16bf bv = lds_b_frag(R2, 136, nt * 16, kc * 32, lane);
        oacc[nt] = wmma_bf16(pf, bv, oacc[nt]);
      }
    }
  }
  const int cl = lane & 15;
  const int rb = qb * 128 + wq0 + ((lane & 16) ? 8 : 0);
#pragma unroll
  for (int v = 0; v < 8; ++v) {
    const float inv = 1.f / lrow[v];
#pragma unroll
    for (int nt = 0; nt < 4; ++nt)
      Op[(size_t)(rb + v) * Hd + nt * 16 + cl] = oacc[nt][v] * inv;
  }
}

// ---------------------------------------------------------------------------
__global__ __launch_bounds__(256) void k_ln(
    const float* __restrict__ A, const float* __restrict__ gam,
    const float* __restrict__ bet, float* __restrict__ out)
{
  const int row = blockIdx.x;
  const int tid = threadIdx.x;
  __shared__ float red[256];
  const float* a = A + (size_t)row * Hd;
  float x[4], s = 0.f;
#pragma unroll
  for (int j = 0; j < 4; ++j) { x[j] = a[tid + j * 256]; s += x[j]; }
  red[tid] = s; __syncthreads();
  for (int o = 128; o > 0; o >>= 1) { if (tid < o) red[tid] += red[tid + o]; __syncthreads(); }
  const float mean = red[0] * (1.f / 1024.f);
  __syncthreads();
  float v = 0.f;
#pragma unroll
  for (int j = 0; j < 4; ++j) { const float d = x[j] - mean; v += d * d; }
  red[tid] = v; __syncthreads();
  for (int o = 128; o > 0; o >>= 1) { if (tid < o) red[tid] += red[tid + o]; __syncthreads(); }
  const float inv = rsqrtf(red[0] * (1.f / 1024.f) + 1e-5f);
  float* op = out + (size_t)row * Hd;
#pragma unroll
  for (int j = 0; j < 4; ++j) {
    const int c = tid + j * 256;
    op[c] = (x[j] - mean) * inv * gam[c] + bet[c];
  }
}

__global__ __launch_bounds__(256) void k_maxpool(
    const float* __restrict__ in, float* __restrict__ out, int tsh)
{
  const size_t i = (size_t)blockIdx.x * 256 + threadIdx.x;  // over B*Tout*H
  const int h = (int)(i & 1023);
  const size_t r = i >> 10;
  const size_t b = r >> tsh, t = r & ((1u << tsh) - 1);
  const float* p = in + ((b << (tsh + 1)) + 2 * t) * Hd + h;
  out[i] = fmaxf(fmaxf(p[0], p[Hd]), 0.f);
}

__global__ __launch_bounds__(256) void k_add(
    const float* __restrict__ a, const float* __restrict__ b,
    float* __restrict__ out, size_t n)
{
  const size_t i = (size_t)blockIdx.x * 256 + threadIdx.x;
  if (i < n) out[i] = a[i] + b[i];
}

// ---------------------------------------------------------------------------
extern "C" void kernel_launch(void* const* d_in, const int* in_sizes, int n_in,
                              void* d_out_v, int out_size, void* d_ws, size_t ws_size,
                              hipStream_t stream) {
  (void)in_sizes; (void)n_in; (void)out_size; (void)ws_size;
  const float* x       = (const float*)d_in[0];
  const float* emb_w   = (const float*)d_in[1];
  const float* emb_b   = (const float*)d_in[2];
  const float* enc_qkvo= (const float*)d_in[3];
  const float* enc_ln  = (const float*)d_in[4];
  const float* enc_w1  = (const float*)d_in[5];
  const float* enc_b1  = (const float*)d_in[6];
  const float* enc_w2  = (const float*)d_in[7];
  const float* enc_b2  = (const float*)d_in[8];
  const float* conv0_w = (const float*)d_in[9];
  const float* conv0_b = (const float*)d_in[10];
  const float* dq0     = (const float*)d_in[11];
  const float* dln0    = (const float*)d_in[12];
  const float* datt1   = (const float*)d_in[13];
  const float* datt2   = (const float*)d_in[14];
  const float* dln     = (const float*)d_in[15];
  const float* dw1     = (const float*)d_in[16];
  const float* db1     = (const float*)d_in[17];
  const float* dw2     = (const float*)d_in[18];
  const float* db2     = (const float*)d_in[19];
  const float* ddw     = (const float*)d_in[20];
  const float* ddb     = (const float*)d_in[21];
  float* out = (float*)d_out_v;
  float* ws  = (float*)d_ws;

  const size_t MU = 1048576ull;        // 1M floats = H*H
  float* X   = ws;                     // 4 MU
  float* T1  = ws + 4 * MU;            // 4 MU
  float* T2  = ws + 8 * MU;            // 4 MU
  float* X2b = ws + 12 * MU;           // 4 MU
  float* Y2b = ws + 16 * MU;           // 4 MU
  float* Fh  = ws + 20 * MU;           // 16 MU (aliases attn bufs)
  float* Qb  = Fh;
  float* Kb  = Fh + 4 * MU;
  float* Vb  = Fh + 8 * MU;
  float* Ob  = Fh + 12 * MU;

  float* E[4] = { out, out + 4 * MU, out + 8 * MU, out + 10 * MU };
  float* D[4] = { out + 11 * MU, out + 12 * MU, out + 14 * MU, out + 18 * MU };

  auto gemm_nn = [&](const float* A, const float* W, float* C, int M, int N, int K) {
    k_gemm_t<false, false, 0><<<dim3(N / 128, M / 128), 256, 0, stream>>>(
        A, W, nullptr, nullptr, C, M, N, K);
  };
  auto gemm_res = [&](const float* A, const float* W, const float* R, float* C,
                      int M, int N, int K) {
    k_gemm_t<false, true, 0><<<dim3(N / 128, M / 128), 256, 0, stream>>>(
        A, W, nullptr, R, C, M, N, K);
  };
  auto gemm_brelu = [&](const float* A, const float* W, const float* bias, float* C,
                        int M, int N, int K) {
    k_gemm_t<true, false, 1><<<dim3(N / 128, M / 128), 256, 0, stream>>>(
        A, W, bias, nullptr, C, M, N, K);
  };
  auto gemm_bres = [&](const float* A, const float* W, const float* bias, const float* R,
                       float* C, int M, int N, int K) {
    k_gemm_t<true, true, 0><<<dim3(N / 128, M / 128), 256, 0, stream>>>(
        A, W, bias, R, C, M, N, K);
  };
  auto lnorm = [&](const float* A, const float* gb, float* O, int M) {
    k_ln<<<M, 256, 0, stream>>>(A, gb, gb + 1024, O);
  };

  // ---- embedding conv + ReLU: (B,DIN,T) -> X (B*T, H)
  k_conv3_t<0><<<dim3(8, 32), 256, 0, stream>>>(x, emb_w, emb_b, X, 1024, 10, 2304, 1024, 1);

  // ---- encoder
  int T = 1024;
  for (int i = 0; i < 4; ++i) {
    const int M = 4 * T;
    const float* Wq = enc_qkvo + (size_t)i * 4 * MU;
    gemm_nn(X, Wq,          Qb, M, Hd, Hd);
    gemm_nn(X, Wq + MU,     Kb, M, Hd, Hd);
    gemm_nn(X, Wq + 2 * MU, Vb, M, Hd, Hd);
    k_attn<<<dim3(T / 128, NHd, Bd), 256, 0, stream>>>(Qb, Kb, Vb, Ob, T, T, 0.125f, 1);
    gemm_res(Ob, Wq + 3 * MU, X, T1, M, Hd, Hd);
    lnorm(T1, enc_ln + (size_t)(i * 2 + 0) * 2048, T2, M);
    gemm_brelu(T2, enc_w1 + (size_t)i * Hd * Fd, enc_b1 + (size_t)i * Fd, Fh, M, Fd, Hd);
    gemm_bres(Fh, enc_w2 + (size_t)i * Fd * Hd, enc_b2 + (size_t)i * Hd, T2, T1, M, Hd, Fd);
    lnorm(T1, enc_ln + (size_t)(i * 2 + 1) * 2048, E[i], M);
    if (i == 0) {
      k_conv3_t<1><<<dim3(8, 32), 256, 0, stream>>>(E[0], conv0_w, conv0_b, X,
                                                    1024, 10, 1024, 1024, 1);
    } else if (i < 3) {
      const int Tn = T / 2;
      k_maxpool<<<(4 * Tn * Hd) / 256, 256, 0, stream>>>(E[i], X, __builtin_ctz(Tn));
      T = Tn;
    }
  }

  // ---- decoder block 0 (self-attention, T=256, no mask)
  {
    const int Td = 256, M = 1024;
    const float* Xd = E[3];
    gemm_nn(Xd, dq0,          Qb, M, Hd, Hd);
    gemm_nn(Xd, dq0 + MU,     Kb, M, Hd, Hd);
    gemm_nn(Xd, dq0 + 2 * MU, Vb, M, Hd, Hd);
    k_attn<<<dim3(2, NHd, Bd), 256, 0, stream>>>(Qb, Kb, Vb, Ob, Td, Td, 0.125f, 0);
    gemm_res(Ob, dq0 + 3 * MU, Xd, T1, M, Hd, Hd);
    lnorm(T1, dln0, D[0], M);
  }

  // ---- decoder upsampling blocks
  int Tf = 256;
  for (int i = 0; i < 3; ++i) {
    const int s  = (i == 2) ? 1 : 2;
    const int To = Tf * s;
    const int M  = 4 * To;
    const float* fe = E[2 - i];
    k_conv3_t<2><<<dim3(8, M / 128), 256, 0, stream>>>(D[i], ddw + (size_t)i * 3 * MU,
                                                       ddb + (size_t)i * Hd, X,
                                                       Tf, __builtin_ctz(To), Hd, Hd, s);
    const float* Wa1 = datt1 + (size_t)i * 4 * MU;
    const float* Wa2 = datt2 + (size_t)i * 4 * MU;
    // attn1: q = fe, kv = X(fd)
    gemm_nn(fe, Wa1,          Qb, M, Hd, Hd);
    gemm_nn(X,  Wa1 + MU,     Kb, M, Hd, Hd);
    gemm_nn(X,  Wa1 + 2 * MU, Vb, M, Hd, Hd);
    k_attn<<<dim3(To / 128, NHd, Bd), 256, 0, stream>>>(Qb, Kb, Vb, Ob, To, To, 0.125f, 0);
    gemm_res(Ob, Wa1 + 3 * MU, fe, T1, M, Hd, Hd);
    lnorm(T1, dln + (size_t)(i * 3 + 0) * 2048, X2b, M);
    // attn2: q = X(fd), kv = fe
    gemm_nn(X,  Wa2,          Qb, M, Hd, Hd);
    gemm_nn(fe, Wa2 + MU,     Kb, M, Hd, Hd);
    gemm_nn(fe, Wa2 + 2 * MU, Vb, M, Hd, Hd);
    k_attn<<<dim3(To / 128, NHd, Bd), 256, 0, stream>>>(Qb, Kb, Vb, Ob, To, To, 0.125f, 0);
    gemm_res(Ob, Wa2 + 3 * MU, X, T1, M, Hd, Hd);
    lnorm(T1, dln + (size_t)(i * 3 + 1) * 2048, Y2b, M);
    // Zd = X2 + Y2 ; FFN(+residual Zd) ; LN -> D[i+1]
    k_add<<<(int)(((size_t)M * Hd) / 256), 256, 0, stream>>>(X2b, Y2b, T2, (size_t)M * Hd);
    gemm_brelu(T2, dw1 + (size_t)i * Hd * Fd, db1 + (size_t)i * Fd, Fh, M, Fd, Hd);
    gemm_bres(Fh, dw2 + (size_t)i * Fd * Hd, db2 + (size_t)i * Hd, T2, T1, M, Hd, Fd);
    lnorm(T1, dln + (size_t)(i * 3 + 2) * 2048, D[i + 1], M);
    Tf = To;
  }
}
